// MultiAttention_46591805227247
// MI455X (gfx1250) — matrix operand order repaired and checked by reading, not compiled, not run on hardware
//
#include <hip/hip_runtime.h>
#include <hip/hip_bf16.h>

// Problem constants (from reference setup_inputs)
#define T_Q 2048
#define S_K 2048
#define NB 4
#define DM 1024
#define NH 16
#define DH 64
#define MROWS (T_Q * NB) // 8192 flattened (t, n) rows

typedef __attribute__((ext_vector_type(16))) __bf16 v16bf;
typedef __attribute__((ext_vector_type(8)))  float  v8f;

union BF16x16 { uint4 u[2]; v16bf v; };
union BF16x8  { uint4 u;    __bf16 h[8]; };

// ---------------------------------------------------------------------------
// Fragment loaders following the CDNA5 WMMA VGPR layouts (ISA 7.12.2).
// A (16x32 bf16, MxK): lane = {kh,m}; VGPR0..3 hold K = kh*8 + 0..7,
//                      VGPR4..7 hold K = 16 + kh*8 + 0..7  -> two 16B loads.
// B (32x16 bf16, KxN): lane = {kh,n}; VGPR0..3 hold K = kh*8 + 0..7,
//                      VGPR4..7 hold K = 16 + kh*8 + 0..7  -> two 16B loads.
// ---------------------------------------------------------------------------
__device__ inline v16bf load_a_frag(const __bf16* base, int ld, int lane) {
  int m = lane & 15, kh = lane >> 4;
  const __bf16* p = base + (size_t)m * ld + kh * 8;
  BF16x16 r;
  r.u[0] = *(const uint4*)(p);
  r.u[1] = *(const uint4*)(p + 16);
  return r.v;
}

__device__ inline v16bf load_b_frag(const __bf16* base, int ld, int lane) {
  int nn = lane & 15, kh = lane >> 4;
  const __bf16* p = base + (size_t)nn * ld + kh * 8;
  BF16x16 r;
  r.u[0] = *(const uint4*)(p);
  r.u[1] = *(const uint4*)(p + 16);
  return r.v;
}

// ---------------------------------------------------------------------------
// f32 -> bf16 bulk conversion: 8 elements/thread, 16B packed store.
// ---------------------------------------------------------------------------
__global__ void __launch_bounds__(256)
cvt_f32_bf16(const float* __restrict__ src, __bf16* __restrict__ dst) {
  size_t i = ((size_t)blockIdx.x * 256 + threadIdx.x) * 8;
  float4 a = *(const float4*)(src + i);
  float4 b = *(const float4*)(src + i + 4);
  BF16x8 o;
  o.h[0] = (__bf16)a.x; o.h[1] = (__bf16)a.y;
  o.h[2] = (__bf16)a.z; o.h[3] = (__bf16)a.w;
  o.h[4] = (__bf16)b.x; o.h[5] = (__bf16)b.y;
  o.h[6] = (__bf16)b.z; o.h[7] = (__bf16)b.w;
  *(uint4*)(dst + i) = o.u;
}

// key_padding_mask (bool8) -> additive logit bias in exp2 domain (0 / -3e38).
__global__ void __launch_bounds__(256)
cvt_mask_bias(const unsigned char* __restrict__ m, float* __restrict__ bias) {
  int i = blockIdx.x * 256 + threadIdx.x;
  bias[i] = m[i] ? -3.0e38f : 0.0f;
}

// ---------------------------------------------------------------------------
// GEMM: C[M,Nc] = A[M,K] @ W[Nc,K]^T + bias.  One wave per 32x64 tile:
// 2 A-fragments share each of 4 B-fragments (8 wmma per K-step of 32).
// ---------------------------------------------------------------------------
template <bool OUT_BF16>
__global__ void __launch_bounds__(256)
gemm_wmma(const __bf16* __restrict__ A, const __bf16* __restrict__ W,
          const float* __restrict__ bias, void* __restrict__ Cout,
          int M, int Nc, int K) {
  int lane = threadIdx.x & 31, wave = threadIdx.x >> 5;
  int tiles_n = Nc >> 6;
  int gid = blockIdx.x * 8 + wave;
  int tm = gid / tiles_n;              // 32-row tile index
  int tn = (gid - tm * tiles_n) << 6;  // 64-col tile start
  if (tm * 32 >= M) return;

  v8f acc[2][4];
  v8f zero = {};
#pragma unroll
  for (int i = 0; i < 2; ++i)
#pragma unroll
    for (int c = 0; c < 4; ++c) acc[i][c] = zero;

  const __bf16* Abase = A + (size_t)tm * 32 * K;
  for (int k0 = 0; k0 < K; k0 += 32) {
    v16bf va0 = load_a_frag(Abase + k0, K, lane);
    v16bf va1 = load_a_frag(Abase + (size_t)16 * K + k0, K, lane);
#pragma unroll
    for (int c = 0; c < 4; ++c) {
      v16bf vb = load_b_frag(W + (size_t)(tn + c * 16) * K + k0, K, lane);
      acc[0][c] = __builtin_amdgcn_wmma_f32_16x16x32_bf16(
          false, va0, false, vb, (short)0, acc[0][c], false, false);
      acc[1][c] = __builtin_amdgcn_wmma_f32_16x16x32_bf16(
          false, va1, false, vb, (short)0, acc[1][c], false, false);
    }
  }

  int hi = lane >> 4, lo = lane & 15;
#pragma unroll
  for (int c = 0; c < 4; ++c) {
    int col = tn + c * 16 + lo;
    float bv = bias[col];
#pragma unroll
    for (int i = 0; i < 2; ++i) {
#pragma unroll
      for (int r = 0; r < 8; ++r) {
        int row = tm * 32 + i * 16 + r + 8 * hi;
        float val = acc[i][c][r] + bv;
        if (OUT_BF16)
          ((__bf16*)Cout)[(size_t)row * Nc + col] = (__bf16)val;
        else
          ((float*)Cout)[(size_t)row * Nc + col] = val;
      }
    }
  }
}

// ---------------------------------------------------------------------------
// Flash attention, transposed-score formulation, 32 queries per wave.
// Block = 8 waves sharing one (batch n, head h); each wave owns two 16-query
// tiles and streams S in blocks of 32.  S^T[s,q] = K_tile x Q^T puts the
// softmax reduction along each lane's VGPRs (one shfl_xor(16) per stat per
// q-tile).  K/V tiles in LDS are shared by both q-tiles: 16 wmma per S-block
// against one async K copy + one V transpose-store.  Softmax runs in the
// exp2 domain (scale*log2e folded into one fma; mask enters as an additive
// precomputed bias), so exp2f lowers directly to v_exp_f32.
// ---------------------------------------------------------------------------
__global__ void __launch_bounds__(256)
flash_attn(const __bf16* __restrict__ Q, const __bf16* __restrict__ Kb,
           const __bf16* __restrict__ Vb, const float* __restrict__ mbias,
           __bf16* __restrict__ O) {
  __shared__ __bf16 sK[32 * 64];        // K tile, row-major (s, d)
  __shared__ __bf16 sVt[64 * 32];       // V tile, transposed (d, s)
  __shared__ __bf16 sP[8][32 * 32];     // per-wave P^T tiles stored as (q, s)

  int lane = threadIdx.x & 31, wave = threadIdx.x >> 5;
  int hi = lane >> 4, lo = lane & 15;
  int n = blockIdx.y >> 4;              // batch
  int h = blockIdx.y & 15;              // head
  int t0 = blockIdx.x * 256 + wave * 32;
  const float SCALE2 = 0.125f * 1.44269504089f;  // (1/sqrt(dh)) * log2(e)

  // Loop-invariant Q^T B-fragments: 2 q-tiles x 2 k-chunks of dh.
  v16bf bQ[2][2];
#pragma unroll
  for (int qt = 0; qt < 2; ++qt) {
    const __bf16* Qbase = Q + ((size_t)(t0 + qt * 16) * NB + n) * DM + h * DH;
#pragma unroll
    for (int c = 0; c < 2; ++c)
      bQ[qt][c] = load_b_frag(Qbase + c * 32, NB * DM, lane);
  }

  v8f acc[2][4];                        // O^T per q-tile: lane col q, rows d
  v8f zero = {};
#pragma unroll
  for (int qt = 0; qt < 2; ++qt)
#pragma unroll
    for (int c = 0; c < 4; ++c) acc[qt][c] = zero;
  float mrun[2] = {-3.0e38f, -3.0e38f};
  float L[2] = {0.f, 0.f};

  // Per-thread staging indices: 256 thr x 16B covers one 32x64 bf16 tile.
  int s_l = threadIdx.x >> 3;
  int off = (threadIdx.x & 7) * 8;
  unsigned ldsK = (unsigned)(uintptr_t)(&sK[s_l * 64 + off]);
  const float* mb = mbias + (size_t)n * S_K;

  for (int sb = 0; sb < S_K; sb += 32) {
    size_t g = ((size_t)(sb + s_l) * NB + n) * DM + h * DH + off;

    // K tile: async global->LDS copy (ASYNCcnt-tracked DMA into LDS).
    {
      const __bf16* gk = Kb + g;
      asm volatile("global_load_async_to_lds_b128 %0, %1, off"
                   :: "v"(ldsK), "v"(gk) : "memory");
    }
    // V tile: load through registers, store transposed (d-major).
    {
      BF16x8 vv; vv.u = *(const uint4*)(Vb + g);
#pragma unroll
      for (int e = 0; e < 8; ++e) sVt[(off + e) * 32 + s_l] = vv.h[e];
    }
    asm volatile("s_wait_asynccnt 0x0" ::: "memory");
    __syncthreads();

    // Scores^T: K-tile A-fragments shared by both q-tiles (8 wmma).
    // D layout: lane holds column q, rows s_local = r + 8*hi (+16j).
    v8f st[2][2];
#pragma unroll
    for (int qt = 0; qt < 2; ++qt)
#pragma unroll
      for (int j = 0; j < 2; ++j) st[qt][j] = zero;
#pragma unroll
    for (int j = 0; j < 2; ++j) {
#pragma unroll
      for (int c = 0; c < 2; ++c) {
        v16bf ak = load_a_frag(&sK[(j * 16) * 64 + c * 32], 64, lane);
#pragma unroll
        for (int qt = 0; qt < 2; ++qt)
          st[qt][j] = __builtin_amdgcn_wmma_f32_16x16x32_bf16(
              false, ak, false, bQ[qt][c], (short)0, st[qt][j], false, false);
      }
    }

    // Additive mask bias (depends only on s -> shared by both q-tiles).
    float bb[2][8];
#pragma unroll
    for (int j = 0; j < 2; ++j) {
      float4 b0 = *(const float4*)(mb + sb + j * 16 + 8 * hi);
      float4 b1 = *(const float4*)(mb + sb + j * 16 + 8 * hi + 4);
      bb[j][0] = b0.x; bb[j][1] = b0.y; bb[j][2] = b0.z; bb[j][3] = b0.w;
      bb[j][4] = b1.x; bb[j][5] = b1.y; bb[j][6] = b1.z; bb[j][7] = b1.w;
    }

    // Online softmax (exp2 domain) + P^T pack, per q-tile.
#pragma unroll
    for (int qt = 0; qt < 2; ++qt) {
      float tmax = -3.0e38f;
#pragma unroll
      for (int j = 0; j < 2; ++j)
#pragma unroll
        for (int r = 0; r < 8; ++r) {
          float v = fmaf(st[qt][j][r], SCALE2, bb[j][r]);
          st[qt][j][r] = v;
          tmax = fmaxf(tmax, v);
        }
      tmax = fmaxf(tmax, __shfl_xor(tmax, 16, 32));
      float mn = fmaxf(mrun[qt], tmax);
      float alpha = exp2f(mrun[qt] - mn);
      float sum = 0.f;
#pragma unroll
      for (int j = 0; j < 2; ++j)
#pragma unroll
        for (int r = 0; r < 8; ++r) {
          float e = exp2f(st[qt][j][r] - mn);
          st[qt][j][r] = e;
          sum += e;
        }
      sum += __shfl_xor(sum, 16, 32);
      L[qt] = L[qt] * alpha + sum;
      mrun[qt] = mn;
#pragma unroll
      for (int c = 0; c < 4; ++c)
#pragma unroll
        for (int r = 0; r < 8; ++r) acc[qt][c][r] *= alpha;

      // Pack P^T into LDS as (q, s) rows: one b128 store per j per lane.
#pragma unroll
      for (int j = 0; j < 2; ++j) {
        BF16x8 pk;
#pragma unroll
        for (int r = 0; r < 8; ++r) pk.h[r] = (__bf16)st[qt][j][r];
        *(uint4*)&sP[wave][(qt * 16 + lo) * 32 + j * 16 + 8 * hi] = pk.u;
      }
    }
    // Same-wave DS ordering guarantees visibility for the reads below.

    // O^T += V^T(64x32) x P^T(32x16): V^T A-frags shared by both q-tiles.
    v16bf bP[2];
#pragma unroll
    for (int qt = 0; qt < 2; ++qt)
      bP[qt] = load_b_frag(&sP[wave][(qt * 16) * 32], 32, lane);
#pragma unroll
    for (int c = 0; c < 4; ++c) {
      v16bf av = load_a_frag(&sVt[(c * 16) * 32], 32, lane);
#pragma unroll
      for (int qt = 0; qt < 2; ++qt)
        acc[qt][c] = __builtin_amdgcn_wmma_f32_16x16x32_bf16(
            false, av, false, bP[qt], (short)0, acc[qt][c], false, false);
    }
    __syncthreads();   // protect sK/sVt before next stage
  }

  // Normalize and store O: lane owns query column -> packed b128 stores.
#pragma unroll
  for (int qt = 0; qt < 2; ++qt) {
    float inv = 1.0f / L[qt];
    __bf16* orow = O + ((size_t)(t0 + qt * 16 + lo) * NB + n) * DM + h * DH;
#pragma unroll
    for (int c = 0; c < 4; ++c) {
      BF16x8 ov;
#pragma unroll
      for (int r = 0; r < 8; ++r) ov.h[r] = (__bf16)(acc[qt][c][r] * inv);
      *(uint4*)(orow + c * 16 + 8 * hi) = ov.u;
    }
  }
}

// ---------------------------------------------------------------------------
extern "C" void kernel_launch(void* const* d_in, const int* in_sizes, int n_in,
                              void* d_out, int out_size, void* d_ws, size_t ws_size,
                              hipStream_t stream) {
  const float* x_q  = (const float*)d_in[0];
  const float* x_k  = (const float*)d_in[1];
  const float* x_v  = (const float*)d_in[2];
  const unsigned char* mask = (const unsigned char*)d_in[3];
  const float* w_in = (const float*)d_in[4];
  const float* b_in = (const float*)d_in[5];
  const float* w_o  = (const float*)d_in[6];
  const float* b_o  = (const float*)d_in[7];
  float* out = (float*)d_out;

  // Workspace carve-up (bf16 buffers + f32 mask bias), ~120 MiB.
  char* p = (char*)d_ws;
  __bf16* Xq   = (__bf16*)p; p += (size_t)MROWS * DM * 2;
  __bf16* Xk   = (__bf16*)p; p += (size_t)MROWS * DM * 2;
  __bf16* Xv   = (__bf16*)p; p += (size_t)MROWS * DM * 2;
  __bf16* Wqkv = (__bf16*)p; p += (size_t)3 * DM * DM * 2;
  __bf16* Wo   = (__bf16*)p; p += (size_t)DM * DM * 2;
  __bf16* Qb   = (__bf16*)p; p += (size_t)MROWS * DM * 2;
  __bf16* Kb   = (__bf16*)p; p += (size_t)MROWS * DM * 2;
  __bf16* Vb   = (__bf16*)p; p += (size_t)MROWS * DM * 2;
  __bf16* Ob   = (__bf16*)p; p += (size_t)MROWS * DM * 2;
  float*  Mb   = (float*)p;  p += (size_t)NB * S_K * 4;
  (void)in_sizes; (void)n_in; (void)out_size; (void)ws_size;

  // 1) fp32 -> bf16 conversions + mask bias precompute.
  cvt_f32_bf16<<<(MROWS * DM) / 2048, 256, 0, stream>>>(x_q, Xq);
  cvt_f32_bf16<<<(MROWS * DM) / 2048, 256, 0, stream>>>(x_k, Xk);
  cvt_f32_bf16<<<(MROWS * DM) / 2048, 256, 0, stream>>>(x_v, Xv);
  cvt_f32_bf16<<<(3 * DM * DM) / 2048, 256, 0, stream>>>(w_in, Wqkv);
  cvt_f32_bf16<<<(DM * DM) / 2048, 256, 0, stream>>>(w_o, Wo);
  cvt_mask_bias<<<(NB * S_K) / 256, 256, 0, stream>>>(mask, Mb);

  // 2) QKV projections: y = x @ W^T + b  (rows of in_proj_weight per torch).
  int gblocks = (MROWS / 32) * (DM / 64) / 8;  // 512
  gemm_wmma<true><<<gblocks, 256, 0, stream>>>(Xq, Wqkv,                       b_in,          Qb, MROWS, DM, DM);
  gemm_wmma<true><<<gblocks, 256, 0, stream>>>(Xk, Wqkv + (size_t)DM * DM,     b_in + DM,     Kb, MROWS, DM, DM);
  gemm_wmma<true><<<gblocks, 256, 0, stream>>>(Xv, Wqkv + (size_t)2 * DM * DM, b_in + 2 * DM, Vb, MROWS, DM, DM);

  // 3) Flash attention per (batch, head): grid = (T/256 q-chunks, NB*NH).
  flash_attn<<<dim3(T_Q / 256, NB * NH), 256, 0, stream>>>(Qb, Kb, Vb, Mb, Ob);

  // 4) Output projection -> fp32 result.
  gemm_wmma<false><<<gblocks, 256, 0, stream>>>(Ob, Wo, b_o, out, MROWS, DM, DM);
}
